// seq2seq_76347338653777
// MI455X (gfx1250) — matrix operand order repaired and checked by reading, not compiled, not run on hardware
//
#include <hip/hip_runtime.h>
#include <hip/hip_bf16.h>

// Problem constants (match reference)
#define kV   32000
#define kE   512
#define kH   1024
#define kS   64
#define kT   64
#define kBsz 32
#define kPAD 1
#define kNEG (-100000.0f)

typedef __attribute__((ext_vector_type(16))) __bf16        v16bf;
typedef __attribute__((ext_vector_type(8)))  float         v8f;
typedef __attribute__((ext_vector_type(4)))  unsigned int  v4u;

// ---------------------------------------------------------------- helpers ---

static __device__ __forceinline__ unsigned short f2bfbits(float f) {
  unsigned u = __builtin_bit_cast(unsigned, f);
  return (unsigned short)((u + 0x7FFFu + ((u >> 16) & 1u)) >> 16);  // RNE
}

static __device__ __forceinline__ v8f wmma_bf16(v16bf a, v16bf b, v8f c) {
  // D = A(16x32) * B(32x16) + C, fp32 accumulate
  return __builtin_amdgcn_wmma_f32_16x16x32_bf16(
      /*neg_a=*/false, a, /*neg_b=*/false, b,
      /*c_mod=*/(short)0, c, /*reuse_a=*/false, /*reuse_b=*/false);
}

// A fragment (16x32 bf16). Row M = lane&15 (both halves). Lane (half)
// elements are one contiguous 16-elem run (matches B fragment):
//   e=0..15 -> K = k0 + 16*half + e
// => two 16-byte vector loads, no ALU. arow points at A[row*lda + k0].
static __device__ __forceinline__ v16bf load_a_frag(const unsigned short* __restrict__ arow,
                                                    int half) {
  union { v4u q[2]; v16bf v; } u;
  u.q[0] = *reinterpret_cast<const v4u*>(arow + 16 * half);
  u.q[1] = *reinterpret_cast<const v4u*>(arow + 16 * half + 8);
  return u.v;
}

// B fragment (32x16 bf16), column N = lane&15; lanes 0-15 hold K=0..15
// sequentially, lanes 16-31 K=16..31 => one contiguous 16-elem run.
// wrow points at W[n*K + k0 + 16*half].
static __device__ __forceinline__ v16bf load_b_frag(const unsigned short* __restrict__ wrow) {
  union { v4u q[2]; v16bf v; } u;
  u.q[0] = *reinterpret_cast<const v4u*>(wrow);
  u.q[1] = *reinterpret_cast<const v4u*>(wrow + 8);
  return u.v;
}

static __device__ __forceinline__ float sigmoidf_(float x) {
  return 1.0f / (1.0f + __expf(-x));
}

// ---------------------------------------------------------------- kernels ---

__global__ void zero_kernel(float* __restrict__ p, long long n) {
  long long i = (long long)blockIdx.x * blockDim.x + threadIdx.x;
  if (i < n) p[i] = 0.0f;
}

// fp32 -> bf16 bits (weights; one-off per launch)
__global__ void quantize_bf16_kernel(const float* __restrict__ src,
                                     unsigned short* __restrict__ dst, long long n) {
  long long i = (long long)blockIdx.x * blockDim.x + threadIdx.x;
  if (i < n) dst[i] = f2bfbits(src[i]);
}

// dst[r*kE + k] = bf16(emb[toks[r]*kE + k])   (pre-gathered embeddings)
__global__ void gather_embed_bf16(const float* __restrict__ emb,
                                  const int* __restrict__ toks,
                                  long long total,
                                  unsigned short* __restrict__ dst) {
  long long i = (long long)blockIdx.x * blockDim.x + threadIdx.x;
  if (i < total) {
    const long long r = i / kE;
    const long long k = i - r * kE;
    dst[i] = f2bfbits(emb[(size_t)toks[r] * kE + k]);
  }
}

// Fused GRU cell: h_out = GRU(x, h_in). All GEMM operands pre-converted bf16.
// One wave owns one 16-column slice of H and both 16-row M tiles (B=32).
// Grid: 16 blocks x 128 threads (64 waves == H/16 column tiles).
__global__ __launch_bounds__(128)
void gru_cell_wmma(const unsigned short* __restrict__ Xbf,    // [B, Kx] bf16
                   int Kx,
                   const unsigned short* __restrict__ Wih,    // [3H, Kx] bf16
                   const unsigned short* __restrict__ Whh,    // [3H, H] bf16
                   const float* __restrict__ bih,             // [3H]
                   const float* __restrict__ bhh,             // [3H]
                   const float* __restrict__ h_in,            // [B, H] fp32
                   const unsigned short* __restrict__ h_in_bf,// [B, H] bf16
                   float* __restrict__ h_out,                 // [B, H] fp32
                   unsigned short* __restrict__ h_out_bf,     // [B, H] bf16
                   float* __restrict__ h_out2) {              // optional copy
  const int gw   = (blockIdx.x * blockDim.x + threadIdx.x) >> 5;
  const int n0   = gw * 16;
  const int lane = threadIdx.x & 31;
  const int half = lane >> 4;
  const int ml   = lane & 15;
  const int n    = n0 + ml;

  const v8f zero = {0.f, 0.f, 0.f, 0.f, 0.f, 0.f, 0.f, 0.f};
  v8f acc_rz[2][2];                 // [gate r/z][mtile] shared across gx+gh
  v8f acc_xn[2], acc_hn[2];         // n-gate parts kept separate
#pragma unroll
  for (int g = 0; g < 2; ++g)
#pragma unroll
    for (int mt = 0; mt < 2; ++mt) acc_rz[g][mt] = zero;
  acc_xn[0] = acc_xn[1] = zero;
  acc_hn[0] = acc_hn[1] = zero;

  // ---- gx = x @ Wih^T (gates r,z,n) ----
  for (int k0 = 0; k0 < Kx; k0 += 32) {
    v16bf a0 = load_a_frag(Xbf + (size_t)ml * Kx + k0, half);
    v16bf a1 = load_a_frag(Xbf + (size_t)(16 + ml) * Kx + k0, half);
    v16bf br = load_b_frag(Wih + (size_t)(0 * kH + n) * Kx + k0 + 16 * half);
    v16bf bz = load_b_frag(Wih + (size_t)(1 * kH + n) * Kx + k0 + 16 * half);
    v16bf bn = load_b_frag(Wih + (size_t)(2 * kH + n) * Kx + k0 + 16 * half);
    acc_rz[0][0] = wmma_bf16(a0, br, acc_rz[0][0]);
    acc_rz[0][1] = wmma_bf16(a1, br, acc_rz[0][1]);
    acc_rz[1][0] = wmma_bf16(a0, bz, acc_rz[1][0]);
    acc_rz[1][1] = wmma_bf16(a1, bz, acc_rz[1][1]);
    acc_xn[0]    = wmma_bf16(a0, bn, acc_xn[0]);
    acc_xn[1]    = wmma_bf16(a1, bn, acc_xn[1]);
  }
  // ---- gh = h @ Whh^T (gates r,z,n) ----
  for (int k0 = 0; k0 < kH; k0 += 32) {
    v16bf a0 = load_a_frag(h_in_bf + (size_t)ml * kH + k0, half);
    v16bf a1 = load_a_frag(h_in_bf + (size_t)(16 + ml) * kH + k0, half);
    v16bf br = load_b_frag(Whh + (size_t)(0 * kH + n) * kH + k0 + 16 * half);
    v16bf bz = load_b_frag(Whh + (size_t)(1 * kH + n) * kH + k0 + 16 * half);
    v16bf bn = load_b_frag(Whh + (size_t)(2 * kH + n) * kH + k0 + 16 * half);
    acc_rz[0][0] = wmma_bf16(a0, br, acc_rz[0][0]);
    acc_rz[0][1] = wmma_bf16(a1, br, acc_rz[0][1]);
    acc_rz[1][0] = wmma_bf16(a0, bz, acc_rz[1][0]);
    acc_rz[1][1] = wmma_bf16(a1, bz, acc_rz[1][1]);
    acc_hn[0]    = wmma_bf16(a0, bn, acc_hn[0]);
    acc_hn[1]    = wmma_bf16(a1, bn, acc_hn[1]);
  }

  // ---- epilogue: r,z,n gates + blend ----
  const float brz0 = bih[n]            + bhh[n];
  const float brz1 = bih[kH + n]       + bhh[kH + n];
  const float bxn  = bih[2 * kH + n];
  const float bhn  = bhh[2 * kH + n];
#pragma unroll
  for (int mt = 0; mt < 2; ++mt) {
#pragma unroll
    for (int r = 0; r < 8; ++r) {
      const int m  = mt * 16 + r + 8 * half;   // C/D VGPR layout
      const float rg = sigmoidf_(acc_rz[0][mt][r] + brz0);
      const float zg = sigmoidf_(acc_rz[1][mt][r] + brz1);
      const float ng = tanhf(acc_xn[mt][r] + bxn + rg * (acc_hn[mt][r] + bhn));
      const float hp = h_in[(size_t)m * kH + n];
      const float hv = (1.0f - zg) * ng + zg * hp;
      h_out[(size_t)m * kH + n]    = hv;
      h_out_bf[(size_t)m * kH + n] = f2bfbits(hv);
      if (h_out2) h_out2[(size_t)m * kH + n] = hv;
    }
  }
}

// Generic C[32,N] = act(A[32,K] @ W[N,K]^T + bias), A/W bf16.
// Writes Cf (fp32) and/or Cq (bf16), whichever is non-null. act: 0=none,1=tanh.
__global__ __launch_bounds__(128)
void gemm_wmma_bias(const unsigned short* __restrict__ Aq, int lda,
                    const unsigned short* __restrict__ Wq,
                    const float* __restrict__ bias,
                    int K, int N,
                    float* __restrict__ Cf, unsigned short* __restrict__ Cq,
                    long long ldc, int act) {
  const int gw = (blockIdx.x * blockDim.x + threadIdx.x) >> 5;
  const int n0 = gw * 16;
  if (n0 >= N) return;
  const int lane = threadIdx.x & 31;
  const int half = lane >> 4;
  const int ml   = lane & 15;
  const int n    = n0 + ml;

  const v8f zero = {0.f, 0.f, 0.f, 0.f, 0.f, 0.f, 0.f, 0.f};
  v8f acc0 = zero, acc1 = zero;
  const unsigned short* wrow = Wq + (size_t)n * K + 16 * half;

  for (int k0 = 0; k0 < K; k0 += 32) {
    v16bf a0 = load_a_frag(Aq + (size_t)ml * lda + k0, half);
    v16bf a1 = load_a_frag(Aq + (size_t)(16 + ml) * lda + k0, half);
    v16bf b  = load_b_frag(wrow + k0);
    acc0 = wmma_bf16(a0, b, acc0);
    acc1 = wmma_bf16(a1, b, acc1);
  }

  const float bb = bias[n];
#pragma unroll
  for (int r = 0; r < 8; ++r) {
    const int m0 = r + 8 * half;
    float v0 = acc0[r] + bb;
    float v1 = acc1[r] + bb;
    if (act) { v0 = tanhf(v0); v1 = tanhf(v1); }
    if (Cf) {
      Cf[(size_t)m0 * ldc + n]        = v0;
      Cf[(size_t)(16 + m0) * ldc + n] = v1;
    }
    if (Cq) {
      Cq[(size_t)m0 * ldc + n]        = f2bfbits(v0);
      Cq[(size_t)(16 + m0) * ldc + n] = f2bfbits(v1);
    }
  }
}

// Attention over S=64: tiny VALU kernel. Grid: B blocks x 256 threads.
__global__ __launch_bounds__(256)
void attention_kernel(const float* __restrict__ h1,         // [B,H]
                      const float* __restrict__ enc_out,    // [S,B,H]
                      const int*   __restrict__ src_tokens, // [S,B]
                      float* __restrict__ ctx) {            // [B,H]
  const int b    = blockIdx.x;
  const int tid  = threadIdx.x;
  const int lane = tid & 31;
  const int wave = tid >> 5;
  __shared__ float sc[kS];

  for (int s = wave; s < kS; s += 8) {
    const float* e = enc_out + ((size_t)s * kBsz + b) * kH;
    const float* h = h1 + (size_t)b * kH;
    float p = 0.0f;
    for (int k = lane; k < kH; k += 32) p += h[k] * e[k];
#pragma unroll
    for (int off = 16; off > 0; off >>= 1) p += __shfl_xor(p, off, 32);
    if (lane == 0) {
      const bool ok = (src_tokens[s * kBsz + b] != kPAD);
      sc[s] = ok ? p : kNEG;
    }
  }
  __syncthreads();

  if (wave == 0) {
    float v0 = sc[lane], v1 = sc[lane + 32];
    float mx = fmaxf(v0, v1);
#pragma unroll
    for (int off = 16; off > 0; off >>= 1) mx = fmaxf(mx, __shfl_xor(mx, off, 32));
    const float e0 = __expf(v0 - mx), e1 = __expf(v1 - mx);
    float sm = e0 + e1;
#pragma unroll
    for (int off = 16; off > 0; off >>= 1) sm += __shfl_xor(sm, off, 32);
    const float inv = 1.0f / sm;
    sc[lane]      = e0 * inv;
    sc[lane + 32] = e1 * inv;
  }
  __syncthreads();

  for (int k = tid; k < kH; k += 256) {
    float acc = 0.0f;
    for (int s = 0; s < kS; ++s)
      acc += sc[s] * enc_out[((size_t)s * kBsz + b) * kH + k];
    ctx[(size_t)b * kH + k] = acc;
  }
}

// hcat_bf = bf16([h1 | ctx])  ([B, 2H])
__global__ void concat_bf16_kernel(const float* __restrict__ h1,
                                   const float* __restrict__ ctx,
                                   unsigned short* __restrict__ hcat) {
  const int i = blockIdx.x * blockDim.x + threadIdx.x;
  const int total = kBsz * 2 * kH;
  if (i < total) {
    const int m = i / (2 * kH);
    const int k = i - m * (2 * kH);
    const float v = (k < kH) ? h1[(size_t)m * kH + k] : ctx[(size_t)m * kH + (k - kH)];
    hcat[i] = f2bfbits(v);
  }
}

// ------------------------------------------------------------------ host ---

extern "C" void kernel_launch(void* const* d_in, const int* in_sizes, int n_in,
                              void* d_out, int out_size, void* d_ws, size_t ws_size,
                              hipStream_t stream) {
  (void)in_sizes; (void)n_in; (void)out_size; (void)ws_size;

  const int*   src_tokens = (const int*)d_in[0];
  const int*   tgt_tokens = (const int*)d_in[1];
  const float* enc_emb    = (const float*)d_in[2];
  const float* enc_Wih0   = (const float*)d_in[3];
  const float* enc_Whh0   = (const float*)d_in[4];
  const float* enc_bih0   = (const float*)d_in[5];
  const float* enc_bhh0   = (const float*)d_in[6];
  const float* enc_Wih1   = (const float*)d_in[7];
  const float* enc_Whh1   = (const float*)d_in[8];
  const float* enc_bih1   = (const float*)d_in[9];
  const float* enc_bhh1   = (const float*)d_in[10];
  const float* dec_emb    = (const float*)d_in[11];
  const float* dec_Wih0   = (const float*)d_in[12];
  const float* dec_Whh0   = (const float*)d_in[13];
  const float* dec_bih0   = (const float*)d_in[14];
  const float* dec_bhh0   = (const float*)d_in[15];
  const float* dec_Wih1   = (const float*)d_in[16];
  const float* dec_Whh1   = (const float*)d_in[17];
  const float* dec_bih1   = (const float*)d_in[18];
  const float* dec_bhh1   = (const float*)d_in[19];
  const float* Wc         = (const float*)d_in[20];
  const float* bc         = (const float*)d_in[21];
  const float* Wo         = (const float*)d_in[22];
  const float* bo         = (const float*)d_in[23];

  float* out = (float*)d_out;

  // ---- workspace carve-out: fp32 region then bf16 region (~127 MB) ----
  float* p = (float*)d_ws;
  float* h0a = p; p += kBsz * kH;
  float* h0b = p; p += kBsz * kH;
  float* h1a = p; p += kBsz * kH;
  float* h1b = p; p += kBsz * kH;
  float* enc_out = p; p += (size_t)kS * kBsz * kH;
  float* ctx  = p; p += kBsz * kH;

  unsigned short* q = (unsigned short*)p;
  unsigned short* eWih0q = q; q += (size_t)3 * kH * kE;
  unsigned short* eWhh0q = q; q += (size_t)3 * kH * kH;
  unsigned short* eWih1q = q; q += (size_t)3 * kH * kH;
  unsigned short* eWhh1q = q; q += (size_t)3 * kH * kH;
  unsigned short* dWih0q = q; q += (size_t)3 * kH * kE;
  unsigned short* dWhh0q = q; q += (size_t)3 * kH * kH;
  unsigned short* dWih1q = q; q += (size_t)3 * kH * kH;
  unsigned short* dWhh1q = q; q += (size_t)3 * kH * kH;
  unsigned short* Wcq    = q; q += (size_t)kH * 2 * kH;
  unsigned short* Woq    = q; q += (size_t)kV * kH;
  unsigned short* enc_xq = q; q += (size_t)kS * kBsz * kE;
  unsigned short* dec_xq = q; q += (size_t)(kT - 1) * kBsz * kE;
  unsigned short* h0abf  = q; q += kBsz * kH;
  unsigned short* h0bbf  = q; q += kBsz * kH;
  unsigned short* h1abf  = q; q += kBsz * kH;
  unsigned short* h1bbf  = q; q += kBsz * kH;
  unsigned short* hcatq  = q; q += kBsz * 2 * kH;
  unsigned short* ccq    = q; q += kBsz * kH;

  // ---- one-off: quantize weights, gather+quantize embeddings ----
  {
    struct QJ { const float* s; unsigned short* d; long long n; };
    const QJ jobs[10] = {
      {enc_Wih0, eWih0q, (long long)3 * kH * kE},
      {enc_Whh0, eWhh0q, (long long)3 * kH * kH},
      {enc_Wih1, eWih1q, (long long)3 * kH * kH},
      {enc_Whh1, eWhh1q, (long long)3 * kH * kH},
      {dec_Wih0, dWih0q, (long long)3 * kH * kE},
      {dec_Whh0, dWhh0q, (long long)3 * kH * kH},
      {dec_Wih1, dWih1q, (long long)3 * kH * kH},
      {dec_Whh1, dWhh1q, (long long)3 * kH * kH},
      {Wc,       Wcq,    (long long)kH * 2 * kH},
      {Wo,       Woq,    (long long)kV * kH},
    };
    for (int j = 0; j < 10; ++j)
      quantize_bf16_kernel<<<(unsigned)((jobs[j].n + 255) / 256), 256, 0, stream>>>(
          jobs[j].s, jobs[j].d, jobs[j].n);
  }
  {
    const long long ne = (long long)kS * kBsz * kE;
    gather_embed_bf16<<<(unsigned)((ne + 255) / 256), 256, 0, stream>>>(
        enc_emb, src_tokens, ne, enc_xq);
    const long long nd = (long long)(kT - 1) * kBsz * kE;
    gather_embed_bf16<<<(unsigned)((nd + 255) / 256), 256, 0, stream>>>(
        dec_emb, tgt_tokens, nd, dec_xq);
  }

  // outputs[0] = 0 ; initial hidden = 0 (fp32 and bf16 views)
  zero_kernel<<<((long long)kBsz * kV + 255) / 256, 256, 0, stream>>>(out, (long long)kBsz * kV);
  zero_kernel<<<(kBsz * kH + 255) / 256, 256, 0, stream>>>(h0a, kBsz * kH);
  zero_kernel<<<(kBsz * kH + 255) / 256, 256, 0, stream>>>(h1a, kBsz * kH);
  zero_kernel<<<(kBsz * kH / 2 + 255) / 256, 256, 0, stream>>>((float*)h0abf, kBsz * kH / 2);
  zero_kernel<<<(kBsz * kH / 2 + 255) / 256, 256, 0, stream>>>((float*)h1abf, kBsz * kH / 2);

  float* h0c = h0a; float* h0n = h0b;
  float* h1c = h1a; float* h1n = h1b;
  unsigned short* h0cq = h0abf; unsigned short* h0nq = h0bbf;
  unsigned short* h1cq = h1abf; unsigned short* h1nq = h1bbf;

  // ---------------- encoder ----------------
  for (int t = 0; t < kS; ++t) {
    gru_cell_wmma<<<16, 128, 0, stream>>>(enc_xq + (size_t)t * kBsz * kE, kE,
                                          eWih0q, eWhh0q, enc_bih0, enc_bhh0,
                                          h0c, h0cq, h0n, h0nq, nullptr);
    gru_cell_wmma<<<16, 128, 0, stream>>>(h0nq, kH,
                                          eWih1q, eWhh1q, enc_bih1, enc_bhh1,
                                          h1c, h1cq, h1n, h1nq,
                                          enc_out + (size_t)t * kBsz * kH);
    float* tf; unsigned short* tq;
    tf = h0c; h0c = h0n; h0n = tf;   tq = h0cq; h0cq = h0nq; h0nq = tq;
    tf = h1c; h1c = h1n; h1n = tf;   tq = h1cq; h1cq = h1nq; h1nq = tq;
  }

  // ---------------- decoder (teacher forced) ----------------
  for (int t = 0; t < kT - 1; ++t) {
    gru_cell_wmma<<<16, 128, 0, stream>>>(dec_xq + (size_t)t * kBsz * kE, kE,
                                          dWih0q, dWhh0q, dec_bih0, dec_bhh0,
                                          h0c, h0cq, h0n, h0nq, nullptr);
    gru_cell_wmma<<<16, 128, 0, stream>>>(h0nq, kH,
                                          dWih1q, dWhh1q, dec_bih1, dec_bhh1,
                                          h1c, h1cq, h1n, h1nq, nullptr);
    attention_kernel<<<kBsz, 256, 0, stream>>>(h1n, enc_out, src_tokens, ctx);
    concat_bf16_kernel<<<(kBsz * 2 * kH + 255) / 256, 256, 0, stream>>>(h1n, ctx, hcatq);
    // cc = tanh(hcat @ Wc^T + bc)   (N=1024, K=2048) -> bf16
    gemm_wmma_bias<<<(kH / 16) / 4, 128, 0, stream>>>(hcatq, 2 * kH, Wcq, bc,
                                                      2 * kH, kH,
                                                      nullptr, ccq, (long long)kH, 1);
    // logits = cc @ Wo^T + bo -> outputs[t+1]  (N=32000, K=1024) -> fp32
    gemm_wmma_bias<<<(kV / 16) / 4, 128, 0, stream>>>(ccq, kH, Woq, bo,
                                                      kH, kV,
                                                      out + (size_t)(t + 1) * kBsz * kV,
                                                      nullptr, (long long)kV, 0);
    float* tf; unsigned short* tq;
    tf = h0c; h0c = h0n; h0n = tf;   tq = h0cq; h0cq = h0nq; h0nq = tq;
    tf = h1c; h1c = h1n; h1n = tf;   tq = h1cq; h1cq = h1nq; h1nq = tq;
  }
}
